// CRF_FB_33921651703971
// MI455X (gfx1250) — matrix operand order repaired and checked by reading, not compiled, not run on hardware
//
#include <hip/hip_runtime.h>
#include <hip/hip_bf16.h>
#include <stdint.h>

// ---------------------------------------------------------------------------
// Types for CDNA5 WMMA / TDM
// ---------------------------------------------------------------------------
typedef __attribute__((ext_vector_type(16))) __bf16 v16bf;
typedef __attribute__((ext_vector_type(8)))  __bf16 v8bf;
typedef __attribute__((ext_vector_type(8)))  float  v8f;
typedef unsigned int v4u __attribute__((ext_vector_type(4)));
typedef int          v4i __attribute__((ext_vector_type(4)));
typedef int          v8i __attribute__((ext_vector_type(8)));

#define S_LEN   2050   // T + 2
#define T_LEN   2048
#define E_DIM   1024
#define H_DIM   512
#define G_DIM   2048   // 4H
#define NTAG    12
#define TAG_START 10
#define TAG_STOP  11

// ---------------------------------------------------------------------------
// fp32 -> bf16 (RNE) conversion with zero K-padding and optional row reversal.
// dst is [rows, dst_ld] bf16 (as uint16 bits), src is [rows, src_ld] fp32.
// rev=1: dst row r <- src row rows-1-r  (backward LSTM direction input).
// ---------------------------------------------------------------------------
__global__ void cvt_bf16_pad(const float* __restrict__ src, uint16_t* __restrict__ dst,
                             int rows, int cols, int src_ld, int dst_ld, int rev) {
  int i = blockIdx.x * blockDim.x + threadIdx.x;
  int n = rows * dst_ld;
  if (i >= n) return;
  int r = i / dst_ld, c = i - r * dst_ld;
  uint16_t out = 0;
  if (c < cols) {
    int sr = rev ? (rows - 1 - r) : r;
    float v = src[(size_t)sr * src_ld + c];
    uint32_t u = __float_as_uint(v);
    u += 0x7FFFu + ((u >> 16) & 1u);     // round-to-nearest-even
    out = (uint16_t)(u >> 16);
  }
  dst[i] = out;
}

// ---------------------------------------------------------------------------
// Tensor Data Mover: issue a 2D tile load (tile_dim0=32 elems of 2B per row,
// tile_dim1=128 rows, row stride = stride_elems) into LDS at lds_byte_addr.
// LDS padding: 16B inserted per 64B stored -> LDS row pitch = 80B = LDT elems.
// D# layout per CDNA5 ISA.
// ---------------------------------------------------------------------------
__device__ __forceinline__ void tdm_load_tile(uint32_t lds_byte_addr,
                                              const uint16_t* gptr,
                                              uint32_t rows_rem,
                                              uint32_t stride_elems) {
  uint64_t ga = (uint64_t)(uintptr_t)gptr;
  v4u g0;
  g0[0] = 1u;                                   // count=1, user descriptor
  g0[1] = lds_byte_addr;                        // lds_addr
  g0[2] = (uint32_t)ga;                         // global_addr[31:0]
  g0[3] = (uint32_t)((ga >> 32) & 0x01FFFFFFu)  // global_addr[56:32]
          | (2u << 30);                         // type=2 ("image")
  v8i g1;
  g1[0] = (int)((1u << 16)      // data_size = 1 -> 2 bytes
              | (1u << 20)      // pad_enable
              | (3u << 22)      // pad_interval = 3 -> every 16 DWORDs (64B)
              | (3u << 25));    // pad_amount   = 3 -> 4 DWORDs (16B)
  g1[1] = (int)(32u << 16);                     // tensor_dim0 = 32 (low16 @ bits79:48)
  g1[2] = (int)((rows_rem & 0xFFFFu) << 16);    // tensor_dim1[15:0] @ bits95:80
  g1[3] = (int)(32u << 16);                     // tile_dim0 = 32 @ bits127:112
  g1[4] = 128;                                  // tile_dim1 = 128 @ bits143:128
  g1[5] = (int)stride_elems;                    // tensor_dim0_stride[31:0]
  g1[6] = 0;
  g1[7] = 0;
  v4i z4 = {0, 0, 0, 0};
#if defined(__clang_major__) && (__clang_major__ >= 23)
  v8i z8 = {0, 0, 0, 0, 0, 0, 0, 0};
  __builtin_amdgcn_tensor_load_to_lds(g0, g1, z4, z4, z8, 0);
#else
  __builtin_amdgcn_tensor_load_to_lds(g0, g1, z4, z4, 0);
#endif
}

// ---------------------------------------------------------------------------
// WMMA bf16 GEMM:  C[M,N](f32) = A[M,K] * W[N,K]^T + bias
//   A, W as bf16 (uint16 bits), row-major, K padded to multiple of 32.
// Block = 256 threads (8 waves), tile 128x128, K-step 32.
// Wave (wm 0..3, wn 0..1) owns a 32x64 sub-tile = 2x4 WMMA fragments.
// TDM double-buffered staging: wave 0 DMAs tile k+1 while all waves WMMA tile k.
// ---------------------------------------------------------------------------
#define BM 128
#define BN 128
#define BK 32
#define LDT 40   // LDS row pitch (bf16 elems): 64B data + 16B TDM pad

__global__ __launch_bounds__(256) void gemm_bf16_wmma(
    const uint16_t* __restrict__ A, int lda,
    const uint16_t* __restrict__ B, int ldb,
    float* __restrict__ C, int ldc,
    const float* __restrict__ bias,
    int M, int N, int Kp)
{
  __shared__ uint16_t At[2][BM * LDT];
  __shared__ uint16_t Bt[2][BN * LDT];

  const int tid  = threadIdx.x;
  const int lane = tid & 31;
  const int wave = tid >> 5;
  const int wm   = wave & 3;       // 0..3  (M direction)
  const int wn   = wave >> 2;      // 0..1  (N direction)
  const int l16  = lane & 15;
  const int hi   = lane >> 4;      // 0 or 1

  const int m0 = blockIdx.y * BM;
  const int n0 = blockIdx.x * BN;

  v8f acc[2][4];
  #pragma unroll
  for (int a = 0; a < 2; a++)
    #pragma unroll
    for (int b = 0; b < 4; b++)
      #pragma unroll
      for (int e = 0; e < 8; e++) acc[a][b][e] = 0.f;

  const uint32_t ldsA[2] = { (uint32_t)(uintptr_t)&At[0][0],
                             (uint32_t)(uintptr_t)&At[1][0] };
  const uint32_t ldsB[2] = { (uint32_t)(uintptr_t)&Bt[0][0],
                             (uint32_t)(uintptr_t)&Bt[1][0] };
  const bool loader = (tid < 32);            // wave 0 drives the TDM
  const uint32_t mrem = (uint32_t)(M - m0);
  const uint32_t nrem = (uint32_t)(N - n0);

  // prologue: DMA first K-tile
  if (loader) {
    tdm_load_tile(ldsA[0], A + (size_t)m0 * lda, mrem, (uint32_t)lda);
    tdm_load_tile(ldsB[0], B + (size_t)n0 * ldb, nrem, (uint32_t)ldb);
    __builtin_amdgcn_s_wait_tensorcnt(0);
  }
  __syncthreads();

  int cur = 0;
  for (int k0 = 0; k0 < Kp; k0 += BK) {
    const int nk = k0 + BK;
    // async: DMA next K-tile into the other buffer while we compute
    if (nk < Kp && loader) {
      tdm_load_tile(ldsA[cur ^ 1], A + (size_t)m0 * lda + nk, mrem, (uint32_t)lda);
      tdm_load_tile(ldsB[cur ^ 1], B + (size_t)n0 * ldb + nk, nrem, (uint32_t)ldb);
    }

    const uint16_t* Ac = At[cur];
    const uint16_t* Bc = Bt[cur];

    // ---- build fragments per ISA 7.12.2 layouts ----
    // A 16x32: lane<16 -> M=l16, K {0..7,16..23}; lane>=16 -> K {8..15,24..31}
    v16bf afr[2];
    #pragma unroll
    for (int fm = 0; fm < 2; fm++) {
      const __bf16* ar = (const __bf16*)&Ac[(wm * 32 + fm * 16 + l16) * LDT];
      ((v8bf*)&afr[fm])[0] = *(const v8bf*)(ar + hi * 8);
      ((v8bf*)&afr[fm])[1] = *(const v8bf*)(ar + hi * 8 + 16);
    }
    // B 32x16: lane<16 -> N=l16, K {0..7,16..23}; lane>=16 -> K {8..15,24..31}
    v16bf bfr[4];
    #pragma unroll
    for (int fn = 0; fn < 4; fn++) {
      const __bf16* br = (const __bf16*)&Bc[(wn * 64 + fn * 16 + l16) * LDT];
      ((v8bf*)&bfr[fn])[0] = *(const v8bf*)(br + hi * 8);
      ((v8bf*)&bfr[fn])[1] = *(const v8bf*)(br + hi * 8 + 16);
    }

    #pragma unroll
    for (int fm = 0; fm < 2; fm++)
      #pragma unroll
      for (int fn = 0; fn < 4; fn++)
        acc[fm][fn] = __builtin_amdgcn_wmma_f32_16x16x32_bf16(
            false, afr[fm], false, bfr[fn], (short)0, acc[fm][fn], false, false);

    if (nk < Kp) {
      if (loader) __builtin_amdgcn_s_wait_tensorcnt(0);  // next buffer landed
      __syncthreads();                                   // publish to all waves
      cur ^= 1;
    }
  }

  // ---- store: C/D layout: VGPR r -> M = r (+8 for hi half), N = l16 ----
  #pragma unroll
  for (int fm = 0; fm < 2; fm++) {
    #pragma unroll
    for (int fn = 0; fn < 4; fn++) {
      int n = n0 + wn * 64 + fn * 16 + l16;
      if (n >= N) continue;
      float bv = bias ? bias[n] : 0.f;
      #pragma unroll
      for (int r = 0; r < 8; r++) {
        int m = m0 + wm * 32 + fm * 16 + (hi ? r + 8 : r);
        if (m < M) C[(size_t)m * ldc + n] = acc[fm][fn][r] + bv;
      }
    }
  }
}

// ---------------------------------------------------------------------------
// Sequential LSTM recurrence, one block per direction.
//   xp  : [2][S][2048] precomputed xs@Wih^T + b (direction-d rows in xs order)
//   Whh : [2][2048][512]
//   out : [S][1024]  (fwd -> cols 0..511 at row t; bwd -> cols 512..1023 at S-1-t)
// ---------------------------------------------------------------------------
__global__ __launch_bounds__(1024) void lstm_scan(
    const float* __restrict__ xp, const float* __restrict__ Whh,
    float* __restrict__ out, int S)
{
  __shared__ float h[H_DIM];
  __shared__ float c[H_DIM];
  __shared__ float g[G_DIM];
  const int d   = blockIdx.x;
  const int tid = threadIdx.x;
  const float* xpd = xp + (size_t)d * S * G_DIM;
  const float* W   = Whh + (size_t)d * G_DIM * H_DIM;
  if (tid < H_DIM) { h[tid] = 0.f; c[tid] = 0.f; }
  __syncthreads();

  const float* w0 = W + (size_t)tid * H_DIM;
  const float* w1 = W + (size_t)(tid + 1024) * H_DIM;

  for (int t = 0; t < S; t++) {
    const float* xr = xpd + (size_t)t * G_DIM;
    float a0 = xr[tid];
    float a1 = xr[tid + 1024];
    #pragma unroll 8
    for (int j = 0; j < H_DIM; j++) {
      float hv = h[j];
      a0 = fmaf(w0[j], hv, a0);
      a1 = fmaf(w1[j], hv, a1);
    }
    g[tid] = a0; g[tid + 1024] = a1;
    __syncthreads();
    if (tid < H_DIM) {
      float gi = g[tid], gf = g[H_DIM + tid], gg = g[2 * H_DIM + tid], go = g[3 * H_DIM + tid];
      float si = 1.f / (1.f + expf(-gi));
      float sf = 1.f / (1.f + expf(-gf));
      float so = 1.f / (1.f + expf(-go));
      float cn = sf * c[tid] + si * tanhf(gg);
      float hn = so * tanhf(cn);
      c[tid] = cn; h[tid] = hn;
      int orow = (d == 0) ? t : (S - 1 - t);
      out[(size_t)orow * (2 * H_DIM) + d * H_DIM + tid] = hn;
    }
    __syncthreads();
  }
}

// ---------------------------------------------------------------------------
// Elementwise helpers
// ---------------------------------------------------------------------------
__device__ __forceinline__ float eluf(float x) { return x > 0.f ? x : (expf(x) - 1.f); }

__global__ void elu_residual(const float* __restrict__ y, const float* __restrict__ x,
                             float* __restrict__ o, int n) {
  int i = blockIdx.x * blockDim.x + threadIdx.x;
  if (i < n) o[i] = eluf(y[i]) + x[i];
}
__global__ void elu_inplace(float* __restrict__ y, int n) {
  int i = blockIdx.x * blockDim.x + threadIdx.x;
  if (i < n) y[i] = eluf(y[i]);
}
__global__ void mul_ew(const float* __restrict__ a, const float* __restrict__ b,
                       float* __restrict__ o, int n) {
  int i = blockIdx.x * blockDim.x + threadIdx.x;
  if (i < n) o[i] = a[i] * b[i];
}

// ---------------------------------------------------------------------------
// CRF forward recursions + gold scores -> scalar.  One block, 32 threads.
//   fu [T,12], fp [T+1,144], trans [12,12], tags [T] int32
// ---------------------------------------------------------------------------
__global__ __launch_bounds__(32) void crf_final(
    const float* __restrict__ fu, const float* __restrict__ fp,
    const float* __restrict__ trans, const int* __restrict__ tags,
    float* __restrict__ out)
{
  __shared__ float tr[NTAG * NTAG];
  __shared__ float fv[NTAG], fvn[NTAG];
  __shared__ float res[2];
  const int j = threadIdx.x;
  for (int i = j; i < NTAG * NTAG; i += 32) tr[i] = trans[i];
  if (j < NTAG) fv[j] = (j == TAG_START) ? 0.f : -100.f;
  __syncthreads();

  // ---- unary chain ----
  for (int t = 0; t < T_LEN; t++) {
    if (j < NTAG) {
      float m = -1e30f;
      #pragma unroll
      for (int i = 0; i < NTAG; i++) m = fmaxf(m, fv[i] + tr[i * NTAG + j]);
      float s = 0.f;
      #pragma unroll
      for (int i = 0; i < NTAG; i++) s += expf(fv[i] + tr[i * NTAG + j] - m);
      fvn[j] = m + logf(s) + fu[t * NTAG + j];
    }
    __syncthreads();
    if (j < NTAG) fv[j] = fvn[j];
    __syncthreads();
  }
  if (j == 0) {
    float m = -1e30f;
    for (int i = 0; i < NTAG; i++) m = fmaxf(m, fv[i] + tr[i * NTAG + TAG_STOP]);
    float s = 0.f;
    for (int i = 0; i < NTAG; i++) s += expf(fv[i] + tr[i * NTAG + TAG_STOP] - m);
    res[0] = m + logf(s);                                 // alpha_u
    float sc = tr[TAG_START * NTAG + tags[0]];
    for (int t = 0; t < T_LEN; t++) sc += fu[t * NTAG + tags[t]];
    for (int t = 1; t < T_LEN; t++) sc += tr[tags[t - 1] * NTAG + tags[t]];
    sc += tr[tags[T_LEN - 1] * NTAG + TAG_STOP];
    res[1] = sc;                                          // score_u
  }
  // ---- pairwise chain ----
  if (j < NTAG) fv[j] = 0.f;
  __syncthreads();
  for (int t = 0; t <= T_LEN; t++) {
    if (j < NTAG) {
      const float* ft = fp + (size_t)t * (NTAG * NTAG);
      float m = -1e30f;
      #pragma unroll
      for (int i = 0; i < NTAG; i++) m = fmaxf(m, fv[i] + ft[i * NTAG + j]);
      float s = 0.f;
      #pragma unroll
      for (int i = 0; i < NTAG; i++) s += expf(fv[i] + ft[i * NTAG + j] - m);
      fvn[j] = m + logf(s);
    }
    __syncthreads();
    if (j < NTAG) fv[j] = fvn[j];
    __syncthreads();
  }
  if (j == 0) {
    float m = -1e30f;
    for (int i = 0; i < NTAG; i++) m = fmaxf(m, fv[i]);
    float s = 0.f;
    for (int i = 0; i < NTAG; i++) s += expf(fv[i] - m);
    float alpha_p = m + logf(s);
    float sp = 0.f; int prev = TAG_START;
    for (int t = 0; t <= T_LEN; t++) {
      int nxt = (t < T_LEN) ? tags[t] : TAG_STOP;
      sp += fp[(size_t)t * (NTAG * NTAG) + prev * NTAG + nxt];
      prev = nxt;
    }
    out[0] = (res[0] - res[1]) + (alpha_p - sp);
  }
}

// ---------------------------------------------------------------------------
// Host orchestration
// ---------------------------------------------------------------------------
extern "C" void kernel_launch(void* const* d_in, const int* in_sizes, int n_in,
                              void* d_out, int out_size, void* d_ws, size_t ws_size,
                              hipStream_t stream) {
  const float* embeds = (const float*)d_in[0];
  const int*   tags   = (const int*)d_in[1];
  const float* Wih_a[2] = { (const float*)d_in[2], (const float*)d_in[5] };
  const float* Whh_a[2] = { (const float*)d_in[3], (const float*)d_in[6] };
  const float* b_a[2]   = { (const float*)d_in[4], (const float*)d_in[7] };
  const float* ufc_W = (const float*)d_in[8];
  const float* ufc_b = (const float*)d_in[9];
  const float* h2t_W = (const float*)d_in[10];
  const float* h2t_b = (const float*)d_in[11];
  const float* trans = (const float*)d_in[12];
  const float* U_W = (const float*)d_in[13]; const float* U_b = (const float*)d_in[14];
  const float* V_W = (const float*)d_in[15]; const float* V_b = (const float*)d_in[16];
  const float* P_W = (const float*)d_in[17]; const float* P_b = (const float*)d_in[18];
  const float* pfc_W = (const float*)d_in[19]; const float* pfc_b = (const float*)d_in[20];
  const float* pp_W = (const float*)d_in[21];  const float* pp_b = (const float*)d_in[22];

  // workspace arena
  size_t off = 0;
  char* base = (char*)d_ws;
  auto alloc = [&](size_t bytes) -> void* {
    void* p = base + off; off += (bytes + 255) & ~(size_t)255; return p;
  };
  uint16_t* Ab16 = (uint16_t*)alloc((size_t)S_LEN * 1024 * 2);
  uint16_t* Bb16 = (uint16_t*)alloc((size_t)2048 * 1024 * 2);
  float* xp    = (float*)alloc((size_t)2 * S_LEN * G_DIM * 4);
  float* xA    = (float*)alloc((size_t)S_LEN * 1024 * 4);
  float* lo_u  = (float*)alloc((size_t)S_LEN * 1024 * 4);
  float* lo_p  = (float*)alloc((size_t)S_LEN * 1024 * 4);
  float* y1    = (float*)alloc((size_t)T_LEN * 1024 * 4);
  float* fcout = (float*)alloc((size_t)T_LEN * 1024 * 4);
  float* feats_u = (float*)alloc((size_t)T_LEN * NTAG * 4);
  float* Ubuf  = (float*)alloc((size_t)(T_LEN + 1) * 512 * 4);
  float* Vbuf  = (float*)alloc((size_t)(T_LEN + 1) * 512 * 4);
  float* Hbuf  = (float*)alloc((size_t)(T_LEN + 1) * 512 * 4);
  float* fi    = (float*)alloc((size_t)(T_LEN + 1) * 300 * 4);
  float* fo_y  = (float*)alloc((size_t)(T_LEN + 1) * 300 * 4);
  float* fobuf = (float*)alloc((size_t)(T_LEN + 1) * 300 * 4);
  float* feats_p = (float*)alloc((size_t)(T_LEN + 1) * 144 * 4);

  auto cvt = [&](const float* src, uint16_t* dst, int rows, int cols, int sld, int dld, int rev) {
    int n = rows * dld;
    cvt_bf16_pad<<<(n + 255) / 256, 256, 0, stream>>>(src, dst, rows, cols, sld, dld, rev);
  };
  auto gemm = [&](const uint16_t* A, int lda, const uint16_t* B, int ldb,
                  float* C, int ldc, const float* bias, int M, int N, int Kp) {
    dim3 g((N + BN - 1) / BN, (M + BM - 1) / BM);
    gemm_bf16_wmma<<<g, 256, 0, stream>>>(A, lda, B, ldb, C, ldc, bias, M, N, Kp);
  };

  // -------- BiLSTMs (u then p) --------
  float* lo_out[2] = { lo_u, lo_p };
  for (int L = 0; L < 2; L++) {
    const float* x = embeds;
    for (int l = 0; l < 2; l++) {
      for (int d = 0; d < 2; d++) {
        cvt(x, Ab16, S_LEN, 1024, 1024, 1024, d);   // d=1: reversed sequence
        const float* wih = Wih_a[L] + ((size_t)(l * 2 + d)) * G_DIM * E_DIM;
        const float* bb  = b_a[L]   + ((size_t)(l * 2 + d)) * G_DIM;
        cvt(wih, Bb16, G_DIM, E_DIM, E_DIM, E_DIM, 0);
        gemm(Ab16, 1024, Bb16, 1024, xp + (size_t)d * S_LEN * G_DIM, G_DIM,
             bb, S_LEN, G_DIM, 1024);
      }
      float* dst = (l == 0) ? xA : lo_out[L];
      const float* whh = Whh_a[L] + (size_t)l * 2 * G_DIM * H_DIM;
      lstm_scan<<<dim3(2), dim3(1024), 0, stream>>>(xp, whh, dst, S_LEN);
      x = dst;
    }
  }

  // -------- unary head --------
  cvt(lo_u + 1024, Ab16, T_LEN, 1024, 1024, 1024, 0);         // rows 1..T
  cvt(ufc_W, Bb16, 1024, 1024, 1024, 1024, 0);
  gemm(Ab16, 1024, Bb16, 1024, y1, 1024, ufc_b, T_LEN, 1024, 1024);
  { int n = T_LEN * 1024;
    elu_residual<<<(n + 255) / 256, 256, 0, stream>>>(y1, lo_u + 1024, fcout, n); }
  cvt(fcout, Ab16, T_LEN, 1024, 1024, 1024, 0);
  cvt(h2t_W, Bb16, NTAG, 1024, 1024, 1024, 0);
  gemm(Ab16, 1024, Bb16, 1024, feats_u, NTAG, h2t_b, T_LEN, NTAG, 1024);

  // -------- pairwise head --------
  const int M2 = T_LEN + 1;   // 2049
  cvt(lo_p, Ab16, M2, 1024, 1024, 1024, 0);                   // rows 0..T
  cvt(U_W, Bb16, 512, 1024, 1024, 1024, 0);
  gemm(Ab16, 1024, Bb16, 1024, Ubuf, 512, U_b, M2, 512, 1024);
  cvt(lo_p + 1024, Ab16, M2, 1024, 1024, 1024, 0);            // rows 1..T+1
  cvt(V_W, Bb16, 512, 1024, 1024, 1024, 0);
  gemm(Ab16, 1024, Bb16, 1024, Vbuf, 512, V_b, M2, 512, 1024);
  { int n = M2 * 512;
    mul_ew<<<(n + 255) / 256, 256, 0, stream>>>(Ubuf, Vbuf, Hbuf, n); }
  cvt(Hbuf, Ab16, M2, 512, 512, 512, 0);
  cvt(P_W, Bb16, 300, 512, 512, 512, 0);
  gemm(Ab16, 512, Bb16, 512, fi, 300, P_b, M2, 300, 512);
  { int n = M2 * 300;
    elu_inplace<<<(n + 255) / 256, 256, 0, stream>>>(fi, n); }
  cvt(fi, Ab16, M2, 300, 300, 320, 0);                        // K pad 300->320
  cvt(pfc_W, Bb16, 300, 300, 300, 320, 0);
  gemm(Ab16, 320, Bb16, 320, fo_y, 300, pfc_b, M2, 300, 320);
  { int n = M2 * 300;
    elu_residual<<<(n + 255) / 256, 256, 0, stream>>>(fo_y, fi, fobuf, n); }
  cvt(fobuf, Ab16, M2, 300, 300, 320, 0);
  cvt(pp_W, Bb16, 144, 300, 300, 320, 0);
  gemm(Ab16, 320, Bb16, 320, feats_p, 144, pp_b, M2, 144, 320);

  // -------- CRF chains + scores -> scalar --------
  crf_final<<<1, 32, 0, stream>>>(feats_u, feats_p, trans, tags, (float*)d_out);

  (void)in_sizes; (void)n_in; (void)out_size; (void)ws_size;
}
